// ClusterMemory_24833500906022
// MI455X (gfx1250) — matrix operand order repaired and checked by reading, not compiled, not run on hardware
//
#include <hip/hip_runtime.h>
#include <math.h>

typedef __attribute__((ext_vector_type(16))) _Float16 v16h;
typedef __attribute__((ext_vector_type(8)))  float    v8f;

#define NS 100000          // memory bank rows
#define NF 256             // feature dim
#define NB 256             // batch
#define INVT 20.0f         // 1/TEMP
#define MOM 0.2f
#define SMOOTHF 0.1f
#define TAUW 0.09f
#define NEG_BIG (-1.0e30f)

// exp(logit - 20) in base-2: 2^((dot-1) * K2), K2 = 20*log2(e)
#define K2 28.853900817779268f
#define LN2F 0.6931471805599453f

#define NTILES (NS / 16)    // 6250 n-tiles of 16
#define GBLK 125            // gemm grid
#define TPB (NTILES / GBLK) // 50 tiles per block

#if __has_builtin(__builtin_amdgcn_sched_barrier)
#define SCHED_FENCE() __builtin_amdgcn_sched_barrier(0)
#else
#define SCHED_FENCE()
#endif

// Raw v_exp_f32 (args are bounded in [-58, 0]: no subnormal fixup needed).
#if __has_builtin(__builtin_amdgcn_exp2f)
#define EXP2_RAW(x) __builtin_amdgcn_exp2f(x)
#else
#define EXP2_RAW(x) exp2f(x)
#endif

// ---------------------------------------------------------------------------
// 1) L2-normalize rows of inputs1/inputs2; emit f32 x1 and f16 x1/x2 copies.
__global__ void normalize_rows(const float* __restrict__ in1,
                               const float* __restrict__ in2,
                               float* __restrict__ x1n,
                               _Float16* __restrict__ x1h,
                               _Float16* __restrict__ x2h) {
  __shared__ float s[NB];
  const int row = blockIdx.x;
  const int j   = threadIdx.x;
  const float* src = (row < NB) ? (in1 + row * NF) : (in2 + (row - NB) * NF);
  float v = src[j];
  s[j] = v * v;
  for (int stp = 128; stp > 0; stp >>= 1) {
    __syncthreads();
    if (j < stp) s[j] += s[j + stp];
  }
  __syncthreads();
  float den = fmaxf(sqrtf(s[0]), 1e-12f);
  float xn = v / den;
  if (row < NB) {
    x1n[row * NF + j] = xn;
    x1h[row * NF + j] = (_Float16)xn;
  } else {
    x2h[(row - NB) * NF + j] = (_Float16)xn;
  }
}

// ---------------------------------------------------------------------------
// 1b) One-shot f32 -> f16 conversion of the feature bank (memory-bound;
//     output stays resident in the 192 MB L2 for the GEMM stream).
__global__ void convert_bank(const float* __restrict__ F,
                             _Float16* __restrict__ Fh) {
  const size_t base = ((size_t)blockIdx.x * blockDim.x + threadIdx.x) * 8;
  float4 a = ((const float4*)(F + base))[0];
  float4 b = ((const float4*)(F + base))[1];
  union { _Float16 h[8]; uint4 q; } u;
  u.h[0] = (_Float16)a.x; u.h[1] = (_Float16)a.y;
  u.h[2] = (_Float16)a.z; u.h[3] = (_Float16)a.w;
  u.h[4] = (_Float16)b.x; u.h[5] = (_Float16)b.y;
  u.h[6] = (_Float16)b.z; u.h[7] = (_Float16)b.w;
  *(uint4*)(Fh + base) = u.q;
}

// ---------------------------------------------------------------------------
// 2) d[b] = x1n[b] . F[t_b]; segment softmax weights w[b] (O(B^2), trivial).
__global__ void dots_weights(const float* __restrict__ x1n,
                             const float* __restrict__ F,
                             const int* __restrict__ tg,
                             float* __restrict__ dvec,
                             float* __restrict__ wvec) {
  __shared__ float sd[NB];
  __shared__ int   st[NB];
  const int b = threadIdx.x;
  st[b] = tg[b];
  const float* xr = x1n + b * NF;
  const float* fr = F + (size_t)tg[b] * NF;
  float acc = 0.f;
  #pragma unroll 8
  for (int i = 0; i < NF; ++i) acc += xr[i] * fr[i];
  sd[b] = acc;
  __syncthreads();
  const int tb = st[b];
  float mx = NEG_BIG;
  for (int j = 0; j < NB; ++j)
    if (st[j] == tb) mx = fmaxf(mx, -sd[j] / TAUW);
  float s = 0.f;
  for (int j = 0; j < NB; ++j)
    if (st[j] == tb) s += __expf(-sd[j] / TAUW - mx);
  wvec[b] = __expf(-acc / TAUW - mx) / s;
  dvec[b] = acc;
}

// ---------------------------------------------------------------------------
// 3) Fused dual GEMM (x1.F^T, x2.F^T) with fixed-shift softmax accumulation.
//    Unit-norm operands => |dot| <= 1 => logits in [-20,20]; exp(logit-20)
//    never overflows/underflows in f32, so no online max tracking needed.
//    Block: 512 threads = 16 waves; wave w owns M-tile rows [16w,16w+16).
//    A fragments (both inputs, all 8 K-chunks) stay resident in VGPRs.
//    F16SRC=1: whole B tile (8 chunks, 64 VGPRs) issued as one deep load
//    clause, kept ahead of the WMMA burst by a sched_barrier so the
//    scheduler cannot re-serialize into load->wait0->wmma per chunk.
//    F16SRC=0: fall back to f32 loads + on-the-fly cvt.
template <bool F16SRC>
__global__ __launch_bounds__(512) void gemm_online(
    const _Float16* __restrict__ x1h, const _Float16* __restrict__ x2h,
    const float* __restrict__ F, const _Float16* __restrict__ Fh,
    float* __restrict__ partials) {
  const int lane = threadIdx.x & 31;
  const int wave = threadIdx.x >> 5;
  const int m0   = wave * 16;
  const int mrow = lane & 15;   // A: M = lane % 16
  const int kh   = lane >> 4;   // A: K block of 16

  // --- preload A fragments in contiguous k order (k = 16*kh + s), same as B ---
  v16h a1[8], a2[8];
  {
    const _Float16* r1 = x1h + (m0 + mrow) * NF;
    const _Float16* r2 = x2h + (m0 + mrow) * NF;
    #pragma unroll
    for (int c = 0; c < 8; ++c) {
      union { v16h v; uint4 q[2]; } t;
      t.q[0] = *(const uint4*)(r1 + c * 32 + kh * 16);
      t.q[1] = *(const uint4*)(r1 + c * 32 + 8 + kh * 16);
      a1[c] = t.v;
      t.q[0] = *(const uint4*)(r2 + c * 32 + kh * 16);
      t.q[1] = *(const uint4*)(r2 + c * 32 + 8 + kh * 16);
      a2[c] = t.v;
    }
  }

  // Per accumulator-slot running sums (shift fixed at logit=20).
  float zo[8], zr[8], tt[8];
  #pragma unroll
  for (int r = 0; r < 8; ++r) { zo[r] = 0.f; zr[r] = 0.f; tt[r] = 0.f; }

  const int nl  = lane & 15;  // B: column n = lane % 16
  const int kb  = lane >> 4;  // B: K block of 16
  const int nt0 = blockIdx.x * TPB;

  for (int t = 0; t < TPB; ++t) {
    const int n0 = (nt0 + t) * 16;

    v8f acc1 = {0.f, 0.f, 0.f, 0.f, 0.f, 0.f, 0.f, 0.f};
    v8f acc2 = {0.f, 0.f, 0.f, 0.f, 0.f, 0.f, 0.f, 0.f};

    if (F16SRC) {
      const _Float16* frh = Fh + (size_t)(n0 + nl) * NF + kb * 16;
      if (t + 1 < TPB)
        __builtin_prefetch(Fh + (size_t)(n0 + 16 + nl) * NF + kb * 64, 0, 1);
      // Issue the full B tile as one deep clause (8 live buffers)...
      union { v16h v; uint4 q[2]; } bt[8];
      #pragma unroll
      for (int c = 0; c < 8; ++c) {
        bt[c].q[0] = *(const uint4*)(frh + c * 32);
        bt[c].q[1] = *(const uint4*)(frh + c * 32 + 8);
      }
      // ...and forbid the scheduler from sinking loads past this point,
      // so the WMMA burst sees a progressive s_wait_loadcnt ramp.
      SCHED_FENCE();
      #pragma unroll
      for (int c = 0; c < 8; ++c) {
        acc1 = __builtin_amdgcn_wmma_f32_16x16x32_f16(false, a1[c], false, bt[c].v,
                                                      (short)0, acc1, false, false);
        acc2 = __builtin_amdgcn_wmma_f32_16x16x32_f16(false, a2[c], false, bt[c].v,
                                                      (short)0, acc2, false, false);
      }
    } else {
      const float* fr = F + (size_t)(n0 + nl) * NF + kb * 16;
      if (t + 1 < TPB)
        __builtin_prefetch(F + (size_t)(n0 + 16 + nl) * NF + kb * 128, 0, 1);
      #pragma unroll
      for (int c = 0; c < 8; ++c) {
        const float* p = fr + c * 32;
        float4 q0 = ((const float4*)p)[0];
        float4 q1 = ((const float4*)p)[1];
        float4 q2 = ((const float4*)p)[2];
        float4 q3 = ((const float4*)p)[3];
        v16h bv;
        bv[0]  = (_Float16)q0.x; bv[1]  = (_Float16)q0.y;
        bv[2]  = (_Float16)q0.z; bv[3]  = (_Float16)q0.w;
        bv[4]  = (_Float16)q1.x; bv[5]  = (_Float16)q1.y;
        bv[6]  = (_Float16)q1.z; bv[7]  = (_Float16)q1.w;
        bv[8]  = (_Float16)q2.x; bv[9]  = (_Float16)q2.y;
        bv[10] = (_Float16)q2.z; bv[11] = (_Float16)q2.w;
        bv[12] = (_Float16)q3.x; bv[13] = (_Float16)q3.y;
        bv[14] = (_Float16)q3.z; bv[15] = (_Float16)q3.w;
        acc1 = __builtin_amdgcn_wmma_f32_16x16x32_f16(false, a1[c], false, bv,
                                                      (short)0, acc1, false, false);
        acc2 = __builtin_amdgcn_wmma_f32_16x16x32_f16(false, a2[c], false, bv,
                                                      (short)0, acc2, false, false);
      }
    }

    // eo = 2^((dot-1)*K2) = exp(logit-20); raw v_exp_f32 (args in [-58,0]).
    #pragma unroll
    for (int r = 0; r < 8; ++r) {
      float d1 = acc1[r];
      float d2 = acc2[r];
      float eo = EXP2_RAW(fmaf(d1, K2, -K2));
      float eg = EXP2_RAW(fmaf(d2, K2, -K2));
      zo[r] += eo;
      zr[r] += eg;
      tt[r] = fmaf(eg, d1 * INVT, tt[r]);
    }
  }

  // Cross-lane sum within each 16-lane group (rows m0 + r + 8*kh).
  #pragma unroll
  for (int r = 0; r < 8; ++r) {
    for (int mask = 1; mask < 16; mask <<= 1) {
      zo[r] += __shfl_xor(zo[r], mask);
      zr[r] += __shfl_xor(zr[r], mask);
      tt[r] += __shfl_xor(tt[r], mask);
    }
    if ((lane & 15) == 0) {
      const int row = m0 + r + 8 * kh;
      float* p = partials + ((size_t)blockIdx.x * NB + row) * 3;
      p[0] = zo[r]; p[1] = zr[r]; p[2] = tt[r];
    }
  }
}

// ---------------------------------------------------------------------------
// 4) Sum block partials, compute losses.
//    lse = 20 + ln2 * log2(Zo); soft expectation = T/Zr (shift cancels).
__global__ void combine_stats(const float* __restrict__ partials,
                              const float* __restrict__ dvec,
                              float* __restrict__ out) {
  __shared__ float slc[NB], sls[NB];
  const int b = threadIdx.x;
  float Zo = 0.f, Zr = 0.f, T = 0.f;
  for (int k = 0; k < GBLK; ++k) {
    const float* p = partials + ((size_t)k * NB + b) * 3;
    Zo += p[0];
    Zr += p[1];
    T  += p[2];
  }
  float lse   = 20.0f + LN2F * __log2f(Zo);
  float out_t = dvec[b] * INVT;
  float lc    = lse - out_t;                       // -logp[b, t_b]
  float ls    = (1.f - SMOOTHF) * (lse - T / Zr) + SMOOTHF * lc;
  slc[b] = lc; sls[b] = ls;
  for (int stp = 128; stp > 0; stp >>= 1) {
    __syncthreads();
    if (b < stp) { slc[b] += slc[b + stp]; sls[b] += sls[b + stp]; }
  }
  if (b == 0) { out[0] = slc[0] / NB; out[1] = sls[0] / NB; }
}

// ---------------------------------------------------------------------------
// 5) Overwrite only the <=256 target rows of the copied bank.
__global__ void update_rows(const float* __restrict__ x1n,
                            const float* __restrict__ F,
                            const int* __restrict__ tg,
                            const float* __restrict__ wvec,
                            float* __restrict__ out) {
  __shared__ float s[NB];
  __shared__ int   st[NB];
  const int b = blockIdx.x;
  const int j = threadIdx.x;
  st[j] = tg[j];
  __syncthreads();
  const int t = st[b];
  for (int b2 = 0; b2 < b; ++b2)
    if (st[b2] == t) return;  // uniform: only first occurrence writes row t
  float acc = 0.f;
  for (int b2 = 0; b2 < NB; ++b2)
    if (st[b2] == t) acc += wvec[b2] * x1n[b2 * NF + j];
  float val = MOM * F[(size_t)t * NF + j] + (1.f - MOM) * acc;
  s[j] = val * val;
  for (int stp = 128; stp > 0; stp >>= 1) {
    __syncthreads();
    if (j < stp) s[j] += s[j + stp];
  }
  __syncthreads();
  float den = fmaxf(sqrtf(s[0]), 1e-12f);
  out[2 + (size_t)t * NF + j] = val / den;
}

// ---------------------------------------------------------------------------
extern "C" void kernel_launch(void* const* d_in, const int* in_sizes, int n_in,
                              void* d_out, int out_size, void* d_ws, size_t ws_size,
                              hipStream_t stream) {
  const float* in1 = (const float*)d_in[0];
  const float* in2 = (const float*)d_in[1];
  const float* F   = (const float*)d_in[2];
  const int*   tg  = (const int*)d_in[3];
  float* out = (float*)d_out;
  char*  ws  = (char*)d_ws;

  float*    x1n = (float*)(ws + 0);          // 256*256*4  = 262144 B
  _Float16* x1h = (_Float16*)(ws + 262144);  // 131072 B
  _Float16* x2h = (_Float16*)(ws + 393216);  // 131072 B
  float*    dv  = (float*)(ws + 524288);     // 1024 B
  float*    wv  = (float*)(ws + 525312);     // 1024 B
  float*    pp  = (float*)(ws + 526336);     // 125*256*3*4 = 384000 B
  const size_t fh_off = 1048576;             // 1 MB-aligned region for f16 bank
  _Float16* Fh = (_Float16*)(ws + fh_off);   // 100000*256*2 = 51.2 MB
  const bool use_f16 = ws_size >= fh_off + (size_t)NS * NF * sizeof(_Float16);

  normalize_rows<<<2 * NB, NB, 0, stream>>>(in1, in2, x1n, x1h, x2h);
  dots_weights<<<1, NB, 0, stream>>>(x1n, F, tg, dv, wv);
  if (use_f16) {
    convert_bank<<<(NS * NF / 8) / 512, 512, 0, stream>>>(F, Fh);
    gemm_online<true><<<GBLK, 512, 0, stream>>>(x1h, x2h, F, Fh, pp);
  } else {
    gemm_online<false><<<GBLK, 512, 0, stream>>>(x1h, x2h, F, Fh, pp);
  }
  combine_stats<<<1, NB, 0, stream>>>(pp, dv, out);
  hipMemcpyAsync(out + 2, F, (size_t)NS * NF * sizeof(float),
                 hipMemcpyDeviceToDevice, stream);
  update_rows<<<NB, NB, 0, stream>>>(x1n, F, tg, wv, out);
}
